// SelfAttentionV1_51599737094689
// MI455X (gfx1250) — matrix operand order repaired and checked by reading, not compiled, not run on hardware
//
#include <hip/hip_runtime.h>
#include <hip/hip_bf16.h>

// ---------------------------------------------------------------------------
// Self-attention (B=4, S=2048, D=1024, fp32 in/out) for gfx1250 (MI455X).
// All three GEMMs run on v_wmma_f32_16x16x32_bf16 (bf16 operands, f32 accum).
// Pipeline: convert -> QKV gemm -> scores gemm -> softmax -> PV gemm.
// Each wave computes a 64x32 output tile (8 WMMA per 32-wide k-step, 12 b128
// loads -> 1.5 loads/WMMA); 8 waves/block tile 128x128.
// ---------------------------------------------------------------------------

typedef __attribute__((ext_vector_type(16))) __bf16        v16bf;
typedef __attribute__((ext_vector_type(8)))  float         v8f;
typedef __attribute__((ext_vector_type(8)))  unsigned int  v8u;

#define BATCH 4
#define SEQ   2048
#define DIM   1024

__device__ __forceinline__ v8f wmma_bf16(v16bf a, v16bf b, v8f c) {
  // (neg_a, A, neg_b, B, c_mod, C, reuse_a, reuse_b)
  return __builtin_amdgcn_wmma_f32_16x16x32_bf16(false, a, false, b, (short)0, c,
                                                 false, false);
}

// ---------------------------------------------------------------------------
// 64x32 output block per wave, K-loop in steps of 32.
// A: row-major [M][K] (bf16), base pre-offset to first row of this wave tile.
// B: stored as B^T row-major [N][K] (bf16), base pre-offset to first col row.
// Fragment layouts follow CDNA5 ISA 7.12.2 tables:
//   A (16-bit 16x32): lane(m=lane&15, half=lane>>4), vgpr j holds
//       K = 16*(j>=4) + 8*half + 2*(j&3)  (+0,+1 packed)
//   B (16-bit 32x16): lane(n=lane&15, half=lane>>4), vgpr j holds
//       K = 16*(j>=4) + 8*half + 2*(j&3)  (+0,+1 packed)
// ---------------------------------------------------------------------------
__device__ __forceinline__ void mma64x32(const __bf16* __restrict__ A, int lda,
                                         const __bf16* __restrict__ B, int ldb,
                                         int K, v8f acc[4][2]) {
  const int lane = threadIdx.x & 31;
  const int m16  = lane & 15;
  const int half = lane >> 4;

  for (int k = 0; k < K; k += 32) {
    v8u ua[4], ub[2];
#pragma unroll
    for (int j = 0; j < 8; ++j) {
      const int ka = k + ((j < 4) ? (2 * j + half * 8) : (2 * j + 8 + half * 8));
      const int kb = ka;
#pragma unroll
      for (int i = 0; i < 4; ++i)
        ua[i][j] = *(const unsigned int*)(A + (size_t)(m16 + 16 * i) * lda + ka);
#pragma unroll
      for (int i = 0; i < 2; ++i)
        ub[i][j] = *(const unsigned int*)(B + (size_t)(m16 + 16 * i) * ldb + kb);
    }
    v16bf bf[2];
#pragma unroll
    for (int i = 0; i < 2; ++i) bf[i] = __builtin_bit_cast(v16bf, ub[i]);
#pragma unroll
    for (int i = 0; i < 4; ++i) {
      const v16bf af = __builtin_bit_cast(v16bf, ua[i]);
      acc[i][0] = wmma_bf16(af, bf[0], acc[i][0]);
      acc[i][1] = wmma_bf16(af, bf[1], acc[i][1]);
    }
  }
}

// --------------------------- stage 0: conversions ---------------------------

__global__ __launch_bounds__(256) void f32_to_bf16_kernel(
    const float* __restrict__ in, __bf16* __restrict__ out) {
  const size_t i = ((size_t)blockIdx.x * blockDim.x + threadIdx.x) * 4;
  const float4 v = *(const float4*)(in + i);
  out[i + 0] = (__bf16)v.x;
  out[i + 1] = (__bf16)v.y;
  out[i + 2] = (__bf16)v.z;
  out[i + 3] = (__bf16)v.w;
}

// Wt[c][d] = (bf16) W[d][c], three 1024x1024 matrices selected by blockIdx.z.
__global__ __launch_bounds__(256) void transpose_convert_kernel(
    const float* __restrict__ W0, const float* __restrict__ W1,
    const float* __restrict__ W2, __bf16* __restrict__ T0,
    __bf16* __restrict__ T1, __bf16* __restrict__ T2) {
  __shared__ float tile[32][33];
  const int z = blockIdx.z;
  const float*  W = (z == 0) ? W0 : (z == 1) ? W1 : W2;
  __bf16*       T = (z == 0) ? T0 : (z == 1) ? T1 : T2;
  const int bx = blockIdx.x * 32, by = blockIdx.y * 32;
  const int tx = threadIdx.x, ty = threadIdx.y;  // (32, 8)
#pragma unroll
  for (int i = 0; i < 32; i += 8)
    tile[ty + i][tx] = W[(size_t)(by + ty + i) * DIM + bx + tx];
  __syncthreads();
#pragma unroll
  for (int i = 0; i < 32; i += 8)
    T[(size_t)(bx + ty + i) * DIM + by + tx] = (__bf16)tile[tx][ty + i];
}

// --------------------------- stage 1: QKV GEMM ------------------------------
// C = H[8192,1024] x W[1024,1024] + b.  z=0 -> Q (row major), z=1 -> K (row
// major), z=2 -> V stored transposed as Vt[b][d][t].
// Block tile 128x128: 8 waves as 2(M) x 4(N), each wave 64x32.
__global__ __launch_bounds__(256) void qkv_gemm_kernel(
    const __bf16* __restrict__ Hb, const __bf16* __restrict__ Wqt,
    const __bf16* __restrict__ Wkt, const __bf16* __restrict__ Wvt,
    const float* __restrict__ bq, const float* __restrict__ bk,
    const float* __restrict__ bv, __bf16* __restrict__ Qb,
    __bf16* __restrict__ Kb, __bf16* __restrict__ Vtb) {
  const int wave = threadIdx.x >> 5;
  const int lane = threadIdx.x & 31;
  const int m16 = lane & 15, half = lane >> 4;
  const int mBase = blockIdx.x * 128 + (wave >> 2) * 64;
  const int nBase = blockIdx.y * 128 + (wave & 3) * 32;
  const int z = blockIdx.z;
  const __bf16* Wt  = (z == 0) ? Wqt : (z == 1) ? Wkt : Wvt;
  const float* bias = (z == 0) ? bq : (z == 1) ? bk : bv;

  v8f acc[4][2] = {};
  mma64x32(Hb + (size_t)mBase * DIM, DIM, Wt + (size_t)nBase * DIM, DIM, DIM, acc);

#pragma unroll
  for (int i = 0; i < 4; ++i)
#pragma unroll
    for (int jj = 0; jj < 2; ++jj)
#pragma unroll
      for (int r = 0; r < 8; ++r) {
        const int gm = mBase + i * 16 + r + 8 * half;  // global row in [0,8192)
        const int gc = nBase + jj * 16 + m16;          // global col in [0,1024)
        const float v = acc[i][jj][r] + bias[gc];
        if (z == 2) {
          const int b = gm >> 11, s = gm & (SEQ - 1);
          Vtb[((size_t)b * DIM + gc) * SEQ + s] = (__bf16)v;
        } else if (z == 1) {
          Kb[(size_t)gm * DIM + gc] = (__bf16)v;
        } else {
          Qb[(size_t)gm * DIM + gc] = (__bf16)v;
        }
      }
}

// --------------------------- stage 2: scores GEMM ---------------------------
// scores[b][s][t] = sum_d Q[b][s][d] * K[b][t][d]   (unscaled; softmax scales)
__global__ __launch_bounds__(256) void scores_gemm_kernel(
    const __bf16* __restrict__ Qb, const __bf16* __restrict__ Kb,
    float* __restrict__ scores) {
  const int wave = threadIdx.x >> 5;
  const int lane = threadIdx.x & 31;
  const int m16 = lane & 15, half = lane >> 4;
  const int b = blockIdx.z;
  const int mBase = blockIdx.x * 128 + (wave >> 2) * 64;
  const int nBase = blockIdx.y * 128 + (wave & 3) * 32;

  const __bf16* A = Qb + (size_t)b * SEQ * DIM + (size_t)mBase * DIM;
  const __bf16* B = Kb + (size_t)b * SEQ * DIM + (size_t)nBase * DIM;

  v8f acc[4][2] = {};
  mma64x32(A, DIM, B, DIM, DIM, acc);

  float* out = scores + (size_t)b * SEQ * SEQ;
#pragma unroll
  for (int i = 0; i < 4; ++i)
#pragma unroll
    for (int jj = 0; jj < 2; ++jj)
#pragma unroll
      for (int r = 0; r < 8; ++r) {
        const int gm = mBase + i * 16 + r + 8 * half;
        const int gc = nBase + jj * 16 + m16;
        out[(size_t)gm * SEQ + gc] = acc[i][jj][r];
      }
}

// --------------------------- stage 3: softmax -------------------------------
// One 256-thread block per (b,s) row of 2048 logits. scale = 1/sqrt(1024).
__global__ __launch_bounds__(256) void softmax_kernel(
    const float* __restrict__ scores, __bf16* __restrict__ probs) {
  const float scale = 0.03125f;  // 1/32
  const size_t row = blockIdx.x;
  const float* x = scores + row * SEQ;
  __bf16* p = probs + row * SEQ;
  const int tid = threadIdx.x;
  const int lane = tid & 31, wv = tid >> 5;

  float v[8];
  float m = -3.402823466e38f;
#pragma unroll
  for (int i = 0; i < 8; ++i) {
    v[i] = x[tid + i * 256] * scale;
    m = fmaxf(m, v[i]);
  }
#pragma unroll
  for (int off = 16; off > 0; off >>= 1) m = fmaxf(m, __shfl_xor(m, off, 32));

  __shared__ float smax[8], ssum[8];
  if (lane == 0) smax[wv] = m;
  __syncthreads();
  m = smax[0];
#pragma unroll
  for (int i = 1; i < 8; ++i) m = fmaxf(m, smax[i]);

  float s = 0.f;
#pragma unroll
  for (int i = 0; i < 8; ++i) {
    v[i] = __expf(v[i] - m);
    s += v[i];
  }
#pragma unroll
  for (int off = 16; off > 0; off >>= 1) s += __shfl_xor(s, off, 32);
  if (lane == 0) ssum[wv] = s;
  __syncthreads();
  s = 0.f;
#pragma unroll
  for (int i = 0; i < 8; ++i) s += ssum[i];

  const float inv = 1.f / s;
#pragma unroll
  for (int i = 0; i < 8; ++i) p[tid + i * 256] = (__bf16)(v[i] * inv);
}

// --------------------------- stage 4: out GEMM ------------------------------
// out[b][s][d] = sum_t P[b][s][t] * V[b][t][d], V held transposed Vt[b][d][t].
__global__ __launch_bounds__(256) void out_gemm_kernel(
    const __bf16* __restrict__ probs, const __bf16* __restrict__ Vtb,
    float* __restrict__ out) {
  const int wave = threadIdx.x >> 5;
  const int lane = threadIdx.x & 31;
  const int m16 = lane & 15, half = lane >> 4;
  const int b = blockIdx.z;
  const int mBase = blockIdx.x * 128 + (wave >> 2) * 64;
  const int nBase = blockIdx.y * 128 + (wave & 3) * 32;

  const __bf16* A = probs + (size_t)b * SEQ * SEQ + (size_t)mBase * SEQ;
  const __bf16* B = Vtb + (size_t)b * DIM * SEQ + (size_t)nBase * SEQ;

  v8f acc[4][2] = {};
  mma64x32(A, SEQ, B, SEQ, SEQ, acc);

  float* o = out + (size_t)b * SEQ * DIM;
#pragma unroll
  for (int i = 0; i < 4; ++i)
#pragma unroll
    for (int jj = 0; jj < 2; ++jj)
#pragma unroll
      for (int r = 0; r < 8; ++r) {
        const int gm = mBase + i * 16 + r + 8 * half;
        const int gc = nBase + jj * 16 + m16;
        o[(size_t)gm * DIM + gc] = acc[i][jj][r];
      }
}

// ---------------------------------------------------------------------------

extern "C" void kernel_launch(void* const* d_in, const int* in_sizes, int n_in,
                              void* d_out, int out_size, void* d_ws,
                              size_t ws_size, hipStream_t stream) {
  (void)in_sizes; (void)n_in; (void)out_size; (void)ws_size;

  const float* H  = (const float*)d_in[0];
  const float* Wq = (const float*)d_in[1];
  const float* bq = (const float*)d_in[2];
  const float* Wk = (const float*)d_in[3];
  const float* bk = (const float*)d_in[4];
  const float* Wv = (const float*)d_in[5];
  const float* bv = (const float*)d_in[6];
  float* out = (float*)d_out;

  const size_t BSD = (size_t)BATCH * SEQ * DIM;  // 8,388,608
  const size_t DD  = (size_t)DIM * DIM;          // 1,048,576
  const size_t BSS = (size_t)BATCH * SEQ * SEQ;  // 16,777,216

  char* ws = (char*)d_ws;
  __bf16* Hb     = (__bf16*)ws;  ws += BSD * 2;   // 16 MB
  __bf16* Wqt    = (__bf16*)ws;  ws += DD * 2;    //  2 MB
  __bf16* Wkt    = (__bf16*)ws;  ws += DD * 2;    //  2 MB
  __bf16* Wvt    = (__bf16*)ws;  ws += DD * 2;    //  2 MB
  __bf16* Qb     = (__bf16*)ws;  ws += BSD * 2;   // 16 MB
  __bf16* Kb     = (__bf16*)ws;  ws += BSD * 2;   // 16 MB
  __bf16* Vtb    = (__bf16*)ws;  ws += BSD * 2;   // 16 MB
  float*  scores = (float*)ws;   ws += BSS * 4;   // 64 MB
  __bf16* probs  = (__bf16*)ws;  ws += BSS * 2;   // 32 MB  (total ~166 MB)

  // Stage 0: convert hidden + transpose-convert weights to bf16.
  f32_to_bf16_kernel<<<(unsigned)(BSD / 1024), 256, 0, stream>>>(H, Hb);
  transpose_convert_kernel<<<dim3(DIM / 32, DIM / 32, 3), dim3(32, 8), 0,
                             stream>>>(Wq, Wk, Wv, Wqt, Wkt, Wvt);

  // Stage 1: Q, K, V projections.  M=8192, N=1024, K=1024.
  qkv_gemm_kernel<<<dim3(8192 / 128, 1024 / 128, 3), 256, 0, stream>>>(
      Hb, Wqt, Wkt, Wvt, bq, bk, bv, Qb, Kb, Vtb);

  // Stage 2: scores = Q K^T per batch.  M=2048, N=2048, K=1024.
  scores_gemm_kernel<<<dim3(2048 / 128, 2048 / 128, BATCH), 256, 0, stream>>>(
      Qb, Kb, scores);

  // Stage 3: softmax over each row of 2048 logits.
  softmax_kernel<<<(unsigned)(BATCH * SEQ), 256, 0, stream>>>(scores, probs);

  // Stage 4: out = P V per batch.  M=2048, N=1024, K=2048.
  out_gemm_kernel<<<dim3(2048 / 128, 1024 / 128, BATCH), 256, 0, stream>>>(
      probs, Vtb, out);
}
